// DfDecoderStep_55765855371914
// MI455X (gfx1250) — matrix operand order repaired and checked by reading, not compiled, not run on hardware
//
#include <hip/hip_runtime.h>

// ---- problem constants -----------------------------------------------------
#define BB   4096
#define EMB  256
#define HH   256
#define GG   8
#define FF   96
#define CHN  16
#define OO   5
#define OCC  10
#define KT   5
#define N3H  768          // 3*H
#define NOUT 960          // F*OC
#define BN_EPS 1e-3f

// output flat offsets (floats)
#define COEFS_OFF  0
#define ALPHA_OFF  (BB*OO*FF*2)                 // 3,932,160
#define NSTATE_OFF (ALPHA_OFF + BB)             // 3,936,256
#define NBUF_OFF   (NSTATE_OFF + BB*2*HH)       // 6,033,408

// LDS B-tile row stride (halfs): 264 halfs = 528 B = 132 dwords -> lane bank
// stride of 4 banks for ds_load_b128 (conflict-free within a 16-lane half).
#define BSTRIDE 264

typedef __attribute__((ext_vector_type(16))) _Float16 v16h;
typedef __attribute__((ext_vector_type(8)))  _Float16 v8h;
typedef __attribute__((ext_vector_type(8)))  float    v8f;
typedef __attribute__((ext_vector_type(4)))  int      v4i;

__device__ __forceinline__ float sigmoidf_(float x) { return 1.f / (1.f + expf(-x)); }

// ---- input conversion: f32 -> f16 for emb, h0, h1 --------------------------
__global__ __launch_bounds__(256) void cvt_inputs(const float* __restrict__ emb,
                                                  const float* __restrict__ state,
                                                  _Float16* __restrict__ embH,
                                                  _Float16* __restrict__ h0H,
                                                  _Float16* __restrict__ h1H) {
    int idx = blockIdx.x * 256 + threadIdx.x;           // < B*256
    if (idx >= BB * HH) return;
    int b = idx >> 8, j = idx & 255;
    embH[idx] = (_Float16)emb[idx];
    h0H[idx]  = (_Float16)state[(size_t)b * 512 + j];
    h1H[idx]  = (_Float16)state[(size_t)b * 512 + 256 + j];
}

// ---- weight prep: f16, transposed (N x K), block-diagonal expansion --------
__global__ __launch_bounds__(256) void prep_win(const float* __restrict__ w_in,
                                                _Float16* __restrict__ WT) {   // 256 x 256
    int idx = blockIdx.x * 256 + threadIdx.x;
    if (idx >= HH * EMB) return;
    int n = idx >> 8, k = idx & 255;
    int gn = n >> 5, gk = k >> 5;
    float v = (gn == gk) ? w_in[(gn * 32 + (k & 31)) * 32 + (n & 31)] : 0.f;
    WT[idx] = (_Float16)v;
}

__global__ __launch_bounds__(256) void prep_gruw(const float* __restrict__ k0,
                                                 const float* __restrict__ rk0,
                                                 const float* __restrict__ k1,
                                                 const float* __restrict__ rk1,
                                                 _Float16* __restrict__ k0T,
                                                 _Float16* __restrict__ rk0T,
                                                 _Float16* __restrict__ k1T,
                                                 _Float16* __restrict__ rk1T) {
    int idx = blockIdx.x * 256 + threadIdx.x;           // < 768*256
    if (idx >= N3H * HH) return;
    int n = idx / 256, k = idx & 255;
    size_t src = (size_t)k * N3H + n;                   // weights are K x N row-major
    k0T[idx]  = (_Float16)k0[src];
    rk0T[idx] = (_Float16)rk0[src];
    k1T[idx]  = (_Float16)k1[src];
    rk1T[idx] = (_Float16)rk1[src];
}

__global__ __launch_bounds__(256) void prep_wout(const float* __restrict__ w_out,
                                                 _Float16* __restrict__ WT) {  // 960 x 256
    int idx = blockIdx.x * 256 + threadIdx.x;
    if (idx >= NOUT * HH) return;
    int n = idx / 256, k = idx & 255;
    int g = n / 120, og = n % 120, kg = k >> 5;
    float v = (g == kg) ? w_out[(g * 32 + (k & 31)) * 120 + og] : 0.f;
    WT[idx] = (_Float16)v;
}

// ---- WMMA GEMM: C[M,N] = A[M,256] * Wt[N,256]^T (+bias)(+relu) -------------
// Block = 8 waves, 1 N-tile (16 cols). B tile (16 x 256 halfs, 8 KB) staged
// once into LDS via async-to-LDS (ASYNCcnt path). Each wave computes 4
// M-tiles, reusing the LDS-resident B fragment across 4 WMMAs per k-step.
// Grid: (N/16, M/512).
__global__ __launch_bounds__(256) void gemm_wmma(const _Float16* __restrict__ A,
                                                 const _Float16* __restrict__ Wt,
                                                 const float* __restrict__ bias,
                                                 float* __restrict__ C,
                                                 _Float16* __restrict__ Ch,
                                                 int N, int relu) {
    __shared__ _Float16 sB[16 * BSTRIDE];                // 8448 B

    const int lane = threadIdx.x & 31;
    const int wave = threadIdx.x >> 5;
    const int row0 = blockIdx.y * 512 + wave * 64;       // 4 M-tiles per wave
    const int col0 = blockIdx.x * 16;

    // ---- stage B tile into LDS: thread t copies 32 B (row t>>4, seg t&15) --
    {
        const int r   = threadIdx.x >> 4;                // 0..15
        const int seg = threadIdx.x & 15;                // 0..15 (16 halfs each)
        const _Float16* src = Wt + (size_t)(col0 + r) * 256 + seg * 16;
        _Float16* dst = &sB[r * BSTRIDE + seg * 16];
#if __has_builtin(__builtin_amdgcn_global_load_async_to_lds_b128)
        auto gp = (__attribute__((address_space(1))) v4i*)src;   // AS1, non-const v4i*
        auto lp = (__attribute__((address_space(3))) v4i*)dst;   // AS3 v4i*
        __builtin_amdgcn_global_load_async_to_lds_b128(gp, lp, 0, 0);
        __builtin_amdgcn_global_load_async_to_lds_b128(gp + 1, lp + 1, 0, 0);
#if __has_builtin(__builtin_amdgcn_s_wait_asynccnt)
        __builtin_amdgcn_s_wait_asynccnt(0);
#else
        asm volatile("s_wait_asynccnt 0" ::: "memory");
#endif
#else
        v8h w0 = *(const v8h*)src;
        v8h w1 = *(const v8h*)(src + 8);
        *(v8h*)dst = w0;
        *(v8h*)(dst + 8) = w1;
#endif
    }
    __syncthreads();

    // A fragment (16x32 f16): lane L<16 -> row L, K 0..15; lane L+16 -> K 16..31
    const int ar = lane & 15;
    const int ak = (lane >> 4) * 16;
    // B fragment (32x16 f16): lane n<16 -> col n, K 0..15; lane n+16 -> col n, K 16..31
    const int br = lane & 15;
    const int bk = (lane >> 4) * 16;

    v8f acc[4] = {{}, {}, {}, {}};
#pragma unroll
    for (int kk = 0; kk < 8; ++kk) {
        const int k0 = kk * 32;
        v16h b = *(const v16h*)&sB[br * BSTRIDE + k0 + bk];
#pragma unroll
        for (int m = 0; m < 4; ++m) {
            const _Float16* ap = A + (size_t)(row0 + m * 16 + ar) * 256 + ak + k0;
            v8h alo = *(const v8h*)ap;
            v8h ahi = *(const v8h*)(ap + 8);
            v16h a;
#pragma unroll
            for (int i = 0; i < 8; ++i) { a[i] = alo[i]; a[8 + i] = ahi[i]; }
            acc[m] = __builtin_amdgcn_wmma_f32_16x16x32_f16(false, a, false, b,
                                                            (short)0, acc[m], false, false);
        }
    }

    // D layout: lane<16 -> M base+r (r=0..7), N = lane; lane>=16 -> M base+8+r, N = lane-16
    const int cn = col0 + (lane & 15);
    const float bv = bias ? bias[cn] : 0.f;
#pragma unroll
    for (int m = 0; m < 4; ++m) {
        const int cm0 = row0 + m * 16 + (lane >> 4) * 8;
#pragma unroll
        for (int r = 0; r < 8; ++r) {
            float v = acc[m][r] + bv;
            if (relu) v = fmaxf(v, 0.f);
            size_t idx = (size_t)(cm0 + r) * N + cn;
            C[idx] = v;
            if (Ch) Ch[idx] = (_Float16)v;
        }
    }
}

// ---- GRU gate fusion -------------------------------------------------------
__global__ __launch_bounds__(256) void gru_gate(const float* __restrict__ xs,   // B x 768
                                                const float* __restrict__ hs,   // B x 768
                                                const float* __restrict__ state,// B x 512
                                                int state_off,
                                                float* __restrict__ ns_out,     // B x 512 (d_out)
                                                _Float16* __restrict__ outH,    // B x 256 or null
                                                const float* __restrict__ xinF, // B x 256 (withC)
                                                float* __restrict__ cF,
                                                _Float16* __restrict__ cH,
                                                int withC) {
    int idx = blockIdx.x * 256 + threadIdx.x;            // < B*256
    if (idx >= BB * HH) return;
    int b = idx >> 8, j = idx & 255;
    const float* xr = xs + (size_t)b * N3H;
    const float* hr = hs + (size_t)b * N3H;
    float h    = state[(size_t)b * 512 + state_off + j];
    float z    = sigmoidf_(xr[j]       + hr[j]);
    float r    = sigmoidf_(xr[256 + j] + hr[256 + j]);
    float cand = tanhf(xr[512 + j] + r * hr[512 + j]);
    float o    = z * h + (1.f - z) * cand;
    ns_out[(size_t)b * 512 + state_off + j] = o;
    if (outH) outH[idx] = (_Float16)o;
    if (withC) {
        float c = o + xinF[idx];
        cF[idx] = c;
        cH[idx] = (_Float16)c;
    }
}

// ---- alpha = sigmoid(c . fc_a_w + b) : one wave per row --------------------
__global__ __launch_bounds__(256) void alpha_k(const float* __restrict__ cF,
                                               const float* __restrict__ w,
                                               const float* __restrict__ bias,
                                               float* __restrict__ alpha) {
    int gtid = blockIdx.x * 256 + threadIdx.x;
    int b    = gtid >> 5;
    int lane = threadIdx.x & 31;
    if (b >= BB) return;
    const float* row = cF + (size_t)b * HH;
    float s = 0.f;
#pragma unroll
    for (int j = lane; j < HH; j += 32) s += row[j] * w[j];
#pragma unroll
    for (int off = 16; off; off >>= 1) s += __shfl_down(s, off, 32);
    if (lane == 0) alpha[b] = sigmoidf_(s + bias[0]);
}

// ---- fused grouped conv + pointwise + BN + relu + buffer shift -------------
__global__ __launch_bounds__(128) void conv_bn(const float* __restrict__ c0,     // B x 1 x F x 16
                                               const float* __restrict__ buf,    // B x 4 x F x 16
                                               const float* __restrict__ conv_w, // 5*8*10
                                               const float* __restrict__ pw_w,   // 10 x 10
                                               const float* __restrict__ gamma,
                                               const float* __restrict__ beta,
                                               const float* __restrict__ mean,
                                               const float* __restrict__ var,
                                               float* __restrict__ c0out,        // B x F x 10
                                               float* __restrict__ new_buf) {    // B x 4 x F x 16
    __shared__ float s_cw[400];
    __shared__ float s_pw[100];
    __shared__ float s_bn[40];
    int t = threadIdx.x;
    for (int i = t; i < 400; i += 128) s_cw[i] = conv_w[i];
    for (int i = t; i < 100; i += 128) s_pw[i] = pw_w[i];
    if (t < 10) { s_bn[t] = gamma[t]; s_bn[10 + t] = beta[t]; s_bn[20 + t] = mean[t]; s_bn[30 + t] = var[t]; }
    __syncthreads();

    int idx = blockIdx.x * 128 + t;                      // b*F + f
    if (idx >= BB * FF) return;
    int b = idx / FF, f = idx % FF;

    float x[KT][CHN];
#pragma unroll
    for (int tt = 0; tt < 4; ++tt) {
        const float4* p = (const float4*)(buf + (((size_t)b * 4 + tt) * FF + f) * CHN);
#pragma unroll
        for (int q = 0; q < 4; ++q) {
            float4 v = p[q];
            x[tt][q * 4 + 0] = v.x; x[tt][q * 4 + 1] = v.y;
            x[tt][q * 4 + 2] = v.z; x[tt][q * 4 + 3] = v.w;
        }
    }
    {
        const float4* p = (const float4*)(c0 + ((size_t)b * FF + f) * CHN);
#pragma unroll
        for (int q = 0; q < 4; ++q) {
            float4 v = p[q];
            x[4][q * 4 + 0] = v.x; x[4][q * 4 + 1] = v.y;
            x[4][q * 4 + 2] = v.z; x[4][q * 4 + 3] = v.w;
        }
    }

    // new_buf[b, t, f, :] = x[t+1]   (t = 0..3)
#pragma unroll
    for (int tt = 0; tt < 4; ++tt) {
        float4* d = (float4*)(new_buf + (((size_t)b * 4 + tt) * FF + f) * CHN);
#pragma unroll
        for (int q = 0; q < 4; ++q) {
            float4 v;
            v.x = x[tt + 1][q * 4 + 0]; v.y = x[tt + 1][q * 4 + 1];
            v.z = x[tt + 1][q * 4 + 2]; v.w = x[tt + 1][q * 4 + 3];
            d[q] = v;
        }
    }

    // grouped conv: y[oc] = sum_t sum_i x[t][g*8+i] * cw[t*80 + i*10 + oc], g = oc/5
    float y[OCC];
#pragma unroll
    for (int oc = 0; oc < OCC; ++oc) {
        const int g = oc / 5;
        float acc = 0.f;
#pragma unroll
        for (int tt = 0; tt < KT; ++tt)
#pragma unroll
            for (int i = 0; i < 8; ++i)
                acc += x[tt][g * 8 + i] * s_cw[tt * 80 + i * 10 + oc];
        y[oc] = acc;
    }
    // pointwise + BN + relu
#pragma unroll
    for (int oc = 0; oc < OCC; ++oc) {
        float acc = 0.f;
#pragma unroll
        for (int j = 0; j < OCC; ++j) acc += y[j] * s_pw[j * 10 + oc];
        float inv = rsqrtf(s_bn[30 + oc] + BN_EPS);
        float v = (acc - s_bn[20 + oc]) * inv * s_bn[oc] + s_bn[10 + oc];
        c0out[(size_t)idx * OCC + oc] = fmaxf(v, 0.f);
    }
}

// ---- final: coefs[b,o,0,f,p] = tanh(g5[b,f,oc]) + c0out[b,f,oc], oc=2o+p ---
__global__ __launch_bounds__(256) void combine_k(const float* __restrict__ g5,
                                                 const float* __restrict__ c0out,
                                                 float* __restrict__ coefs) {
    int idx = blockIdx.x * 256 + threadIdx.x;            // < B*960
    if (idx >= BB * NOUT) return;
    float v = tanhf(g5[idx]) + c0out[idx];
    int b = idx / NOUT, rem = idx % NOUT;
    int f = rem / OCC, oc = rem % OCC;
    int o = oc >> 1, p = oc & 1;
    coefs[(((size_t)b * OO + o) * FF + f) * 2 + p] = v;
}

// ---------------------------------------------------------------------------
extern "C" void kernel_launch(void* const* d_in, const int* in_sizes, int n_in,
                              void* d_out, int out_size, void* d_ws, size_t ws_size,
                              hipStream_t stream) {
    const float* emb     = (const float*)d_in[0];
    const float* c0      = (const float*)d_in[1];
    const float* state   = (const float*)d_in[2];
    const float* convbuf = (const float*)d_in[3];
    const float* w_in    = (const float*)d_in[4];
    const float* gru_k0  = (const float*)d_in[5];
    const float* gru_rk0 = (const float*)d_in[6];
    const float* gru_b0  = (const float*)d_in[7];
    const float* gru_k1  = (const float*)d_in[8];
    const float* gru_rk1 = (const float*)d_in[9];
    const float* gru_b1  = (const float*)d_in[10];
    const float* w_out   = (const float*)d_in[11];
    const float* fc_a_w  = (const float*)d_in[12];
    const float* fc_a_b  = (const float*)d_in[13];
    const float* conv_w  = (const float*)d_in[14];
    const float* pw_w    = (const float*)d_in[15];
    const float* bn_g    = (const float*)d_in[16];
    const float* bn_b    = (const float*)d_in[17];
    const float* bn_m    = (const float*)d_in[18];
    const float* bn_v    = (const float*)d_in[19];

    float* out = (float*)d_out;
    float* coefs  = out + COEFS_OFF;
    float* alpha  = out + ALPHA_OFF;
    float* nstate = out + NSTATE_OFF;
    float* nbuf   = out + NBUF_OFF;

    // deterministic workspace carve-up (256B aligned)
    char* wsp = (char*)d_ws;
    auto alloc = [&](size_t bytes) -> void* {
        void* p = (void*)wsp;
        wsp += (bytes + 255) & ~(size_t)255;
        return p;
    };
    _Float16* embH  = (_Float16*)alloc((size_t)BB * HH * 2);
    _Float16* h0H   = (_Float16*)alloc((size_t)BB * HH * 2);
    _Float16* h1H   = (_Float16*)alloc((size_t)BB * HH * 2);
    _Float16* xinH  = (_Float16*)alloc((size_t)BB * HH * 2);
    float*    xinF  = (float*)   alloc((size_t)BB * HH * 4);
    _Float16* out0H = (_Float16*)alloc((size_t)BB * HH * 2);
    _Float16* cH    = (_Float16*)alloc((size_t)BB * HH * 2);
    float*    cF    = (float*)   alloc((size_t)BB * HH * 4);
    float*    xsb   = (float*)   alloc((size_t)BB * N3H * 4);
    float*    hsb   = (float*)   alloc((size_t)BB * N3H * 4);
    float*    g5b   = (float*)   alloc((size_t)BB * NOUT * 4);
    float*    c0out = (float*)   alloc((size_t)BB * NOUT * 4);
    _Float16* WinT  = (_Float16*)alloc((size_t)HH * EMB * 2);
    _Float16* k0T   = (_Float16*)alloc((size_t)N3H * HH * 2);
    _Float16* rk0T  = (_Float16*)alloc((size_t)N3H * HH * 2);
    _Float16* k1T   = (_Float16*)alloc((size_t)N3H * HH * 2);
    _Float16* rk1T  = (_Float16*)alloc((size_t)N3H * HH * 2);
    _Float16* WoutT = (_Float16*)alloc((size_t)NOUT * HH * 2);
    (void)ws_size; (void)n_in; (void)in_sizes; (void)out_size;

    // --- prep ---
    cvt_inputs<<<(BB * HH) / 256, 256, 0, stream>>>(emb, state, embH, h0H, h1H);
    prep_win<<<(HH * EMB) / 256, 256, 0, stream>>>(w_in, WinT);
    prep_gruw<<<(N3H * HH) / 256, 256, 0, stream>>>(gru_k0, gru_rk0, gru_k1, gru_rk1,
                                                    k0T, rk0T, k1T, rk1T);
    prep_wout<<<(NOUT * HH) / 256, 256, 0, stream>>>(w_out, WoutT);

    // --- conv path (independent of GRU chain) ---
    conv_bn<<<(BB * FF) / 128, 128, 0, stream>>>(c0, convbuf, conv_w, pw_w,
                                                 bn_g, bn_b, bn_m, bn_v, c0out, nbuf);

    dim3 blk(256);
    // G0: x_in = relu(emb @ Win)      [N=256]
    gemm_wmma<<<dim3(HH / 16, BB / 512), blk, 0, stream>>>(embH, WinT, nullptr,
                                                           xinF, xinH, HH, 1);
    // G1/G2: GRU0 projections         [N=768]
    gemm_wmma<<<dim3(N3H / 16, BB / 512), blk, 0, stream>>>(xinH, k0T, gru_b0,
                                                            xsb, nullptr, N3H, 0);
    gemm_wmma<<<dim3(N3H / 16, BB / 512), blk, 0, stream>>>(h0H, rk0T, gru_b0 + N3H,
                                                            hsb, nullptr, N3H, 0);
    gru_gate<<<(BB * HH) / 256, 256, 0, stream>>>(xsb, hsb, state, 0, nstate,
                                                  out0H, nullptr, nullptr, nullptr, 0);
    // G3/G4: GRU1 projections (reuse xsb/hsb)
    gemm_wmma<<<dim3(N3H / 16, BB / 512), blk, 0, stream>>>(out0H, k1T, gru_b1,
                                                            xsb, nullptr, N3H, 0);
    gemm_wmma<<<dim3(N3H / 16, BB / 512), blk, 0, stream>>>(h1H, rk1T, gru_b1 + N3H,
                                                            hsb, nullptr, N3H, 0);
    gru_gate<<<(BB * HH) / 256, 256, 0, stream>>>(xsb, hsb, state, HH, nstate,
                                                  nullptr, xinF, cF, cH, 1);
    // alpha
    alpha_k<<<(BB * 32) / 256, 256, 0, stream>>>(cF, fc_a_w, fc_a_b, alpha);
    // G5: grouped out projection      [N=960]
    gemm_wmma<<<dim3(NOUT / 16, BB / 512), blk, 0, stream>>>(cH, WoutT, nullptr,
                                                             g5b, nullptr, NOUT, 0);
    // final coefs
    combine_k<<<(BB * NOUT) / 256, 256, 0, stream>>>(g5b, c0out, coefs);
}
